// GenerativeWorldModel_40235253629006
// MI455X (gfx1250) — matrix operand order repaired and checked by reading, not compiled, not run on hardware
//
#include <hip/hip_runtime.h>
#include <hip/hip_bf16.h>
#include <math.h>

typedef __attribute__((ext_vector_type(16))) __bf16 v16bf;
typedef __attribute__((ext_vector_type(8)))  float  v8f;

#define BB   2048
#define TT   50
#define NN   23
#define FFEA 4
#define HH   64
#define FUT  10
#define G3   192   // 3*H

__device__ __forceinline__ float sigm(float x)  { return 1.0f / (1.0f + __expf(-x)); }
__device__ __forceinline__ float reluf(float x) { return x > 0.0f ? x : 0.0f; }

// One workgroup per batch element; 8 wave32s.
// Wave wv owns M-tile tm=wv>>2 and gate-column block q=wv&3:
//   N-tiles {q, q+4, q+8} == (r, z, n) gate tiles at identical (row, col%64).
// r/z gates accumulate gi+gh in ONE wmma accumulator chain; only the n gate
// keeps gi_n / gh_n separate (r multiplies gh_n before the add).
__global__ __launch_bounds__(256) __attribute__((amdgpu_waves_per_eu(4)))
void stgcn_gru_kernel(const float* __restrict__ x_seq,   // (B,T,N,F)
                      const float* __restrict__ adj_seq, // (B,T,N,N)
                      const float* __restrict__ W_gcn,   // (H,F)
                      const float* __restrict__ b_gcn,   // (H)
                      const float* __restrict__ W_ih,    // (3H,H)
                      const float* __restrict__ W_hh,    // (3H,H)
                      const float* __restrict__ b_ih,    // (3H)
                      const float* __restrict__ b_hh,    // (3H)
                      const float* __restrict__ W_d1,    // (H/2,H)
                      const float* __restrict__ b_d1,    // (H/2)
                      const float* __restrict__ W_d2,    // (F,H/2)
                      const float* __restrict__ b_d2,    // (F)
                      float* __restrict__ out)           // (B,FUT,N,F)
{
    const int tid  = threadIdx.x;
    const int lane = tid & 31;
    const int wv   = tid >> 5;          // wave id 0..7
    const int hi16 = (lane >> 4) & 1;   // upper half-wave
    const int l15  = lane & 15;
    const int b    = blockIdx.x;
    const int tm   = wv >> 2;           // M-tile this wave owns
    const int q    = wv & 3;            // gate-column block

    // ---- LDS (~40 KB) ----
    __shared__ __bf16 sAs [32 * 64];      // spatial, row-major (A of gi matmul)
    __shared__ __bf16 sAhB[2][32 * 64];   // h bf16, double-buffered (A of gh matmul)
    __shared__ __bf16 sAdj[32 * 32];      // adj tile, zero padded (K pad)
    __shared__ __bf16 sSupB[64 * 32];     // support TRANSPOSED [h][n] (B of spatial mm)
    __shared__ float  sH  [32 * 64];      // h fp32, written only during future steps
    __shared__ float  sX  [32 * 4];       // decoded x feeding future st_cells
    __shared__ float  sD1 [32 * 32];      // decoder hidden
    __shared__ float  sWg [64 * 4],  sBg[64];
    __shared__ float  sWd1[32 * 64], sBd1[32];
    __shared__ float  sWd2[4 * 32],  sBd2[4];

    // ---- preload small weights to LDS ----
    sWg[tid] = W_gcn[tid];                          // 256 entries, blockDim==256
    if (tid < 64)  sBg[tid]  = b_gcn[tid];
    for (int i = tid; i < 32 * 64; i += 256) sWd1[i] = W_d1[i];
    if (tid < 32)  sBd1[tid] = b_d1[tid];
    if (tid < 128) sWd2[tid] = W_d2[tid];
    if (tid < 4)   sBd2[tid] = b_d2[tid];
    // h0 = 0 (both bf16 buffers)
    for (int i = tid; i < 32 * 64; i += 256) {
        sAhB[0][i] = (__bf16)0.0f;  sAhB[1][i] = (__bf16)0.0f;
    }

    // ---- per-lane step-invariant gate biases (registers) ----
    const int gc = q * 16 + l15;                    // gate column 0..63
    const float brz0 = b_ih[gc]      + b_hh[gc];        // r: bi+bh merged
    const float brz1 = b_ih[64 + gc] + b_hh[64 + gc];   // z: bi+bh merged
    const float bin  = b_ih[128 + gc];
    const float bhn  = b_hh[128 + gc];

    // ---- GRU weight B-fragments resident in registers for all 60 steps ----
    v16bf Bih[3][2], Bhh[3][2];
#pragma unroll
    for (int ti = 0; ti < 3; ++ti) {
        const int j = (q + ti * 4) * 16 + l15;      // row of W == output column
#pragma unroll
        for (int ks = 0; ks < 2; ++ks) {
#pragma unroll
            for (int e = 0; e < 16; ++e) {
                const int k = ks * 32 + hi16 * 16 + e;
                Bih[ti][ks][e] = (__bf16)W_ih[j * HH + k];
                Bhh[ti][ks][e] = (__bf16)W_hh[j * HH + k];
            }
        }
    }

    // ---- h master copy in registers: this wave's 16x16 patch ----
    float hreg[8];
#pragma unroll
    for (int v = 0; v < 8; ++v) hreg[v] = 0.0f;

    __syncthreads();

    const float* xb = x_seq   + (size_t)b * TT * NN * FFEA;
    const float* ab = adj_seq + (size_t)b * TT * NN * NN;

    for (int step = 0; step < TT + FUT; ++step) {
        const int rb = step & 1, wb = rb ^ 1;       // h bf16 read/write buffers

        // ---- stage 0+1: stage adj (history), build support^T (B-fragment layout) ----
        if (step < TT) {
            for (int i = tid; i < 32 * 32; i += 256) {
                const int r = i >> 5, c = i & 31;
                float v = (r < NN && c < NN) ? ab[step * NN * NN + r * NN + c] : 0.0f;
                sAdj[i] = (__bf16)v;
            }
        }
        {   // thread t -> node n = t>>3, 8 columns starting at (t&7)*8
            const int n  = tid >> 3;
            const int j0 = (tid & 7) * 8;
            float x0 = 0.f, x1 = 0.f, x2 = 0.f, x3 = 0.f;
            if (n < NN) {
                if (step < TT) {
                    const float* xp = xb + step * NN * FFEA + n * FFEA;
                    x0 = xp[0]; x1 = xp[1]; x2 = xp[2]; x3 = xp[3];
                } else {
                    x0 = sX[n*4+0]; x1 = sX[n*4+1]; x2 = sX[n*4+2]; x3 = sX[n*4+3];
                }
            }
#pragma unroll
            for (int j = 0; j < 8; ++j) {
                const int hc = j0 + j;
                float s = 0.0f;
                if (n < NN)
                    s = sBg[hc] + x0*sWg[hc*4+0] + x1*sWg[hc*4+1]
                                + x2*sWg[hc*4+2] + x3*sWg[hc*4+3];
                sSupB[hc * 32 + n] = (__bf16)s;     // transposed: rows>=23 zero => K-pad
            }
        }
        __syncthreads();

        // ---- stage 2: spatial = relu(adj @ support), 1 WMMA tile per wave ----
        {
            const int m = wv >> 2, nt = wv & 3;
            v16bf a, bm;
#pragma unroll
            for (int e = 0; e < 16; ++e) {          // A gather: 16 contiguous bf16
                const int k = hi16 * 16 + e;
                a[e] = sAdj[(m * 16 + l15) * 32 + k];
            }
#pragma unroll
            for (int e = 0; e < 16; ++e)            // B gather: 16 contiguous bf16
                bm[e] = sSupB[(nt * 16 + l15) * 32 + hi16 * 16 + e];
            v8f c = {};
            c = __builtin_amdgcn_wmma_f32_16x16x32_bf16(false, a, false, bm,
                                                        (short)0, c, false, false);
#pragma unroll
            for (int v = 0; v < 8; ++v) {
                const int row = m * 16 + v + hi16 * 8;
                sAs[row * 64 + nt * 16 + l15] = (__bf16)reluf(c[v]);
            }
        }
        __syncthreads();

        // ---- stage 3: gi/gh WMMA + GRU gates fully in registers ----
        {
            v16bf As[2], Ah[2];
#pragma unroll
            for (int ks = 0; ks < 2; ++ks) {
#pragma unroll
                for (int e = 0; e < 16; ++e) {
                    const int k   = ks * 32 + hi16 * 16 + e;
                    const int row = tm * 16 + l15;
                    As[ks][e] = sAs[row * 64 + k];
                    Ah[ks][e] = sAhB[rb][row * 64 + k];
                }
            }
            // r and z: gi+gh fused into one accumulator chain (4 wmma each)
            v8f arz[2];
#pragma unroll
            for (int g = 0; g < 2; ++g) {
                v8f acc = {};
#pragma unroll
                for (int ks = 0; ks < 2; ++ks) {
                    acc = __builtin_amdgcn_wmma_f32_16x16x32_bf16(false, As[ks], false, Bih[g][ks],
                                                                  (short)0, acc, false, false);
                    acc = __builtin_amdgcn_wmma_f32_16x16x32_bf16(false, Ah[ks], false, Bhh[g][ks],
                                                                  (short)0, acc, false, false);
                }
                arz[g] = acc;
            }
            // n: gi_n and gh_n kept separate (r * gh_n before add)
            v8f cin = {}, chn = {};
#pragma unroll
            for (int ks = 0; ks < 2; ++ks) {
                cin = __builtin_amdgcn_wmma_f32_16x16x32_bf16(false, As[ks], false, Bih[2][ks],
                                                              (short)0, cin, false, false);
                chn = __builtin_amdgcn_wmma_f32_16x16x32_bf16(false, Ah[ks], false, Bhh[2][ks],
                                                              (short)0, chn, false, false);
            }
            // gates in registers; padded rows (>=23) carry bounded junk, never read back
#pragma unroll
            for (int v = 0; v < 8; ++v) {
                const int row = tm * 16 + v + hi16 * 8;
                const float r  = sigm(arz[0][v] + brz0);
                const float z  = sigm(arz[1][v] + brz1);
                const float nn = tanhf(cin[v] + bin + r * (chn[v] + bhn));
                const float hn = (1.0f - z) * nn + z * hreg[v];
                hreg[v] = hn;
                sAhB[wb][row * 64 + gc] = (__bf16)hn;       // A of next step's gh
                if (step >= TT) sH[row * 64 + gc] = hn;     // decoder input (fp32)
            }
        }
        __syncthreads();

        // ---- stage 4 (future only): decode h -> prediction, feed next x ----
        if (step >= TT) {
            for (int i = tid; i < NN * 32; i += 256) {
                const int n = i >> 5, j = i & 31;
                float s = sBd1[j];
#pragma unroll 8
                for (int k = 0; k < HH; ++k) s += sH[n * HH + k] * sWd1[j * HH + k];
                sD1[i] = reluf(s);
            }
            __syncthreads();
            if (tid < NN * FFEA) {
                const int n = tid >> 2, f = tid & 3;
                float s = sBd2[f];
#pragma unroll
                for (int k = 0; k < 32; ++k) s += sD1[n * 32 + k] * sWd2[f * 32 + k];
                out[((size_t)b * FUT + (step - TT)) * NN * FFEA + tid] = s;
                sX[tid] = s;                        // feeds next st_cell
            }
            __syncthreads();
        }
    }
}

extern "C" void kernel_launch(void* const* d_in, const int* in_sizes, int n_in,
                              void* d_out, int out_size, void* d_ws, size_t ws_size,
                              hipStream_t stream) {
    (void)in_sizes; (void)n_in; (void)out_size; (void)d_ws; (void)ws_size;
    const float* x_seq   = (const float*)d_in[0];
    const float* adj_seq = (const float*)d_in[1];
    const float* W_gcn   = (const float*)d_in[2];
    const float* b_gcn   = (const float*)d_in[3];
    const float* W_ih    = (const float*)d_in[4];
    const float* W_hh    = (const float*)d_in[5];
    const float* b_ih    = (const float*)d_in[6];
    const float* b_hh    = (const float*)d_in[7];
    const float* W_d1    = (const float*)d_in[8];
    const float* b_d1    = (const float*)d_in[9];
    const float* W_d2    = (const float*)d_in[10];
    const float* b_d2    = (const float*)d_in[11];
    float* out = (float*)d_out;

    stgcn_gru_kernel<<<dim3(BB), dim3(256), 0, stream>>>(
        x_seq, adj_seq, W_gcn, b_gcn, W_ih, W_hh, b_ih, b_hh,
        W_d1, b_d1, W_d2, b_d2, out);
}
